// CapsNet_66357244723585
// MI455X (gfx1250) — matrix operand order repaired and checked by reading, not compiled, not run on hardware
//
#include <hip/hip_runtime.h>

// ---------- types for CDNA5 WMMA (wave32) ----------
typedef __bf16          v16bf __attribute__((ext_vector_type(16)));
typedef unsigned short  v16us __attribute__((ext_vector_type(16)));
typedef float           v8f   __attribute__((ext_vector_type(8)));

__device__ __forceinline__ unsigned short f32_to_bf16(float f) {
  unsigned int u = __builtin_bit_cast(unsigned int, f);
  u += 0x7FFFu + ((u >> 16) & 1u);          // round-to-nearest-even
  return (unsigned short)(u >> 16);
}

// LDS byte offset of a __shared__ object = low 32 bits of its generic
// address (LDS aperture: addr[31:0] is the in-allocation offset).
__device__ __forceinline__ unsigned lds_offset(const void* p) {
  return (unsigned)(size_t)p;
}

// Async global->LDS copy of 16 bytes (tracked by ASYNCcnt, no VGPRs).
__device__ __forceinline__ void async_copy_b128(unsigned lds_off,
                                                const void* gaddr) {
  asm volatile("global_load_async_to_lds_b128 %0, %1, off"
               :: "v"(lds_off), "v"(gaddr) : "memory");
}
__device__ __forceinline__ void async_wait0() {
  asm volatile("s_wait_asynccnt 0x0" ::: "memory");
}

// =====================================================================
// Fused conv3x3(SAME) + bias + ReLU + maxpool2x2 (NCHW, stride 1).
// One block per image: whole input image staged in LDS via async
// global->LDS b128 DMA (every input byte crosses HBM/L2 exactly once).
// Each thread then produces several pooled outputs purely from LDS.
// Next image is prefetched (global_prefetch_b8).
// =====================================================================
template<int CIN, int COUT, int HIN, int WIN>
__global__ void conv_relu_pool(const float* __restrict__ x,
                               const float* __restrict__ w,
                               const float* __restrict__ bias,
                               float* __restrict__ y, int B) {
  constexpr int HOUT = HIN / 2, WOUT = WIN / 2;
  constexpr int NIN = CIN * HIN * WIN;
  constexpr int NW  = COUT * CIN * 9;
  __shared__ float sx[NIN];
  __shared__ float sw[NW];
  __shared__ float sb[COUT];

  const int b = blockIdx.x;
  const float* xb = x + (size_t)b * NIN;

  // prefetch next image while we work on this one
  if (b + 1 < B)
    __builtin_prefetch(xb + NIN + threadIdx.x * 32, 0, 0);

  {  // stage input image via async-to-LDS DMA; weights via normal loads
    unsigned sxa = lds_offset(sx);
    const float4* x4 = (const float4*)xb;
    for (int i = threadIdx.x; i < NIN / 4; i += blockDim.x)
      async_copy_b128(sxa + (unsigned)i * 16u, x4 + i);
    for (int i = threadIdx.x; i < NW; i += blockDim.x) sw[i] = w[i];
    for (int i = threadIdx.x; i < COUT; i += blockDim.x) sb[i] = bias[i];
    async_wait0();
  }
  __syncthreads();

  float* yb = y + (size_t)b * COUT * HOUT * WOUT;
  for (int i = threadIdx.x; i < COUT * HOUT * WOUT; i += blockDim.x) {
    int pw = i % WOUT; int t = i / WOUT;
    int ph = t % HOUT; int co = t / HOUT;

    const float* wf = sw + co * CIN * 9;
    float acc0 = 0.f, acc1 = 0.f, acc2 = 0.f, acc3 = 0.f;
    for (int c = 0; c < CIN; ++c) {
      float p[4][4];
#pragma unroll
      for (int iy = 0; iy < 4; ++iy) {
        int yy = 2 * ph - 1 + iy;
#pragma unroll
        for (int jx = 0; jx < 4; ++jx) {
          int xx = 2 * pw - 1 + jx;
          bool in = (yy >= 0) & (yy < HIN) & (xx >= 0) & (xx < WIN);
          p[iy][jx] = in ? sx[(c * HIN + yy) * WIN + xx] : 0.f;
        }
      }
      const float* wc = wf + c * 9;
      float s0 = 0.f, s1 = 0.f, s2 = 0.f, s3 = 0.f;
#pragma unroll
      for (int ky = 0; ky < 3; ++ky)
#pragma unroll
        for (int kx = 0; kx < 3; ++kx) {
          float wv = wc[ky * 3 + kx];
          s0 = fmaf(p[ky][kx],         wv, s0);
          s1 = fmaf(p[ky][kx + 1],     wv, s1);
          s2 = fmaf(p[ky + 1][kx],     wv, s2);
          s3 = fmaf(p[ky + 1][kx + 1], wv, s3);
        }
      acc0 += s0; acc1 += s1; acc2 += s2; acc3 += s3;
    }
    float bv = sb[co];
    float r0 = fmaxf(acc0 + bv, 0.f), r1 = fmaxf(acc1 + bv, 0.f);
    float r2 = fmaxf(acc2 + bv, 0.f), r3 = fmaxf(acc3 + bv, 0.f);
    yb[i] = fmaxf(fmaxf(r0, r1), fmaxf(r2, r3));
  }
}

// =====================================================================
// u_hat for routing-1:  u[b,n,r,o] = sum_c feats[b,r,c] * W[n,r,c,o]
// For fixed (n,r) this is a GEMM over batch: [B x 32] * [32 x 8].
// Dense packing: TWO capsule types per WMMA (n0 -> cols 0-7, n1 ->
// cols 8-15) => 100% tile utilization. All 8 waves of a block share the
// same r (MT % 8 == 0), so the full per-r weight slice is staged ONCE
// in LDS, pre-swizzled into WMMA lane order as bf16: the inner loop is
// 2x ds_load_b128 + v_wmma per n-pair. A tile held in VGPRs across all
// 16 n-pairs.  feats[b,r,c] = h3[b*2048 + c*64 + r]
// =====================================================================
__global__ void uhat1_wmma(const float* __restrict__ h3,
                           const float* __restrict__ rw1,
                           float* __restrict__ uh, int MT) {
  __shared__ unsigned short sB[16][32][16];        // [np][lane][j], 16 KB
  int wv   = threadIdx.x >> 5;
  int lane = threadIdx.x & 31;
  int r  = (blockIdx.x * 8) / MT;                  // uniform across block
  int mt = (blockIdx.x * 8 + wv) % MT;

  // ---- cooperative stage of B slice for this r (pre-swizzled bf16) ----
  for (int p = threadIdx.x; p < 16 * 32; p += blockDim.x) {
    int ln = p & 31, np = p >> 5;
    int n2 = 2 * np + ((ln >> 3) & 1);             // capsule type
    int o2 = ln & 7, hi2 = ln >> 4;
#pragma unroll
    for (int j = 0; j < 16; ++j) {
      int kb = hi2 * 8 + (j & 7) + ((j >> 3) << 4); // c in [0,32)
      sB[np][ln][j] =
          f32_to_bf16(rw1[((((size_t)n2 * 64 + r) * 32) + kb) * 8 + o2]);
    }
  }

  // ---- A tile (16 batch rows x K=32), loaded once, held in VGPRs ----
  int row = lane & 15;
  int hi  = lane >> 4;
  int b0  = mt * 16 + row;
  v16us au;
#pragma unroll
  for (int j = 0; j < 16; ++j) {
    int ka = hi * 8 + (j & 7) + ((j >> 3) << 4);   // c in [0,32)
    au[j] = f32_to_bf16(h3[(size_t)b0 * 2048 + (size_t)ka * 64 + r]);
  }
  v16bf A = __builtin_bit_cast(v16bf, au);
  __syncthreads();

  const int o    = row & 7;
  const int nsel = row >> 3;

#pragma unroll 2
  for (int np = 0; np < 16; ++np) {
    int n = 2 * np + nsel;
    v16us bu = *(const v16us*)&sB[np][lane][0];    // 2x ds_load_b128
    v8f acc = {};
    acc = __builtin_amdgcn_wmma_f32_16x16x32_bf16(
        false, A, false, __builtin_bit_cast(v16bf, bu),
        (short)0, acc, false, false);

#pragma unroll
    for (int v = 0; v < 8; ++v) {
      int bb = mt * 16 + v + hi * 8;               // batch row
      uh[((((size_t)bb * 32 + n) * 64) + r) * 8 + o] = acc[v];
    }
  }
}

// =====================================================================
// Routing-1 iterations: one wave per (b,n); R=64, O=8.
// u_hat tile (2KB contiguous) staged into LDS via async b128 DMA;
// softmax / reductions via wave32 shuffles.
// =====================================================================
__global__ void routing1_iter(const float* __restrict__ uh,
                              float* __restrict__ v1) {
  __shared__ float suh[8][64][8];                 // 16 KB
  int wv = threadIdx.x >> 5, lane = threadIdx.x & 31;
  int job = blockIdx.x * 8 + wv;                  // b*32 + n
  {
    unsigned ldsa = lds_offset(&suh[wv][0][0]) + (unsigned)lane * 64u;
    const float* g = uh + (size_t)job * 512 + lane * 16;
#pragma unroll
    for (int k = 0; k < 4; ++k)
      async_copy_b128(ldsa + (unsigned)k * 16u, g + k * 4);
    async_wait0();
  }
  __syncthreads();
  float (*U)[8] = suh[wv];

  float b0 = 0.f, b1 = 0.f;                       // logits for r=lane, lane+32
  float vv[8];
  for (int it = 0; it < 3; ++it) {
    float m = fmaxf(b0, b1);
    for (int d = 16; d > 0; d >>= 1) m = fmaxf(m, __shfl_xor(m, d, 32));
    float e0 = __expf(b0 - m), e1 = __expf(b1 - m);
    float s = e0 + e1;
    for (int d = 16; d > 0; d >>= 1) s += __shfl_xor(s, d, 32);
    float inv = 1.f / s;
    float c0 = e0 * inv, c1 = e1 * inv;

    float sj[8];
#pragma unroll
    for (int oo = 0; oo < 8; ++oo) {
      float tt = c0 * U[lane][oo] + c1 * U[lane + 32][oo];
      for (int d = 16; d > 0; d >>= 1) tt += __shfl_xor(tt, d, 32);
      sj[oo] = tt;
    }
    float norm = 0.f;
#pragma unroll
    for (int oo = 0; oo < 8; ++oo) norm += sj[oo] * sj[oo];
    float scale = norm / (1.f + norm) * rsqrtf(norm + 1e-9f);
#pragma unroll
    for (int oo = 0; oo < 8; ++oo) vv[oo] = scale * sj[oo];
    if (it < 2) {
      float d0 = 0.f, d1 = 0.f;
#pragma unroll
      for (int oo = 0; oo < 8; ++oo) {
        d0 += U[lane][oo] * vv[oo];
        d1 += U[lane + 32][oo] * vv[oo];
      }
      b0 += d0; b1 += d1;
    }
  }
  if (lane == 0) {
    float* dst = v1 + (size_t)job * 8;
#pragma unroll
    for (int oo = 0; oo < 8; ++oo) dst[oo] = vv[oo];
  }
}

// =====================================================================
// Routing-2 fully fused: one wave per (b,n); R=32 (one route per lane),
// C=8, O=16. u_hat lives in registers; K=8 makes WMMA 75% padding,
// so scalar FMA is the right choice here.
// =====================================================================
__global__ void routing2_fused(const float* __restrict__ v1,
                               const float* __restrict__ rw2,
                               float* __restrict__ flat) {
  int wv = threadIdx.x >> 5, lane = threadIdx.x & 31;
  int job = blockIdx.x * 8 + wv;                  // b*15 + n
  int n = job % 15, b = job / 15;
  int r = lane;

  float xr[8];
#pragma unroll
  for (int c = 0; c < 8; ++c) xr[c] = v1[((size_t)b * 32 + r) * 8 + c];
  float uhr[16];
#pragma unroll
  for (int o = 0; o < 16; ++o) {
    float s = 0.f;
#pragma unroll
    for (int c = 0; c < 8; ++c)
      s = fmaf(xr[c], rw2[((((size_t)n * 32 + r) * 8) + c) * 16 + o], s);
    uhr[o] = s;
  }

  float bij = 0.f, vv[16];
  for (int it = 0; it < 3; ++it) {
    float m = bij;
    for (int d = 16; d > 0; d >>= 1) m = fmaxf(m, __shfl_xor(m, d, 32));
    float e = __expf(bij - m);
    float s = e;
    for (int d = 16; d > 0; d >>= 1) s += __shfl_xor(s, d, 32);
    float c = e / s;

    float sj[16];
#pragma unroll
    for (int o = 0; o < 16; ++o) {
      float tt = c * uhr[o];
      for (int d = 16; d > 0; d >>= 1) tt += __shfl_xor(tt, d, 32);
      sj[o] = tt;
    }
    float norm = 0.f;
#pragma unroll
    for (int o = 0; o < 16; ++o) norm += sj[o] * sj[o];
    float scale = norm / (1.f + norm) * rsqrtf(norm + 1e-9f);
#pragma unroll
    for (int o = 0; o < 16; ++o) vv[o] = scale * sj[o];
    if (it < 2) {
      float d0 = 0.f;
#pragma unroll
      for (int o = 0; o < 16; ++o) d0 += uhr[o] * vv[o];
      bij += d0;
    }
  }
  if (lane == 0) {
    float* dst = flat + ((size_t)b * 15 + n) * 16;   // flat[b, n*16+o]
#pragma unroll
    for (int o = 0; o < 16; ++o) dst[o] = vv[o];
  }
}

// =====================================================================
// FC: out[b,k] = flat[b,:240] . fc_w[k,:240] + fc_b[k]
// bf16 WMMA GEMM: M=B, K=240->256 (8 chunks of 32), N=15->16.
// fc_w is shared by all waves: staged once per block, pre-swizzled into
// WMMA lane order as bf16 (8 KB). One wave per 16-row batch tile ->
// 8 chained v_wmma_f32_16x16x32_bf16 fed by ds_load_b128.
// =====================================================================
__global__ void fc_wmma(const float* __restrict__ flat,
                        const float* __restrict__ fcw,
                        const float* __restrict__ fcb,
                        float* __restrict__ out) {
  __shared__ unsigned short sB[8][32][16];        // [kk][lane][j], 8 KB
  int wv = threadIdx.x >> 5, lane = threadIdx.x & 31;
  int mt = blockIdx.x * 8 + wv;
  int row = lane & 15, hi = lane >> 4;

  {  // stage fc_w: one (kk,lane) pair per thread
    int kk0 = threadIdx.x >> 5;   // reuse wv: 8 kk chunks x 32 lanes = 256
    int ln  = threadIdx.x & 31;
    int row2 = ln & 15, hi2 = ln >> 4;
#pragma unroll
    for (int j = 0; j < 16; ++j) {
      int kb = kk0 * 32 + hi2 * 8 + (j & 7) + ((j >> 3) << 4);
      float bv = ((row2 < 15) & (kb < 240)) ? fcw[(size_t)row2 * 240 + kb] : 0.f;
      sB[kk0][ln][j] = f32_to_bf16(bv);
    }
  }
  __syncthreads();

  v8f acc = {};
#pragma unroll
  for (int kk = 0; kk < 8; ++kk) {
    v16us au;
#pragma unroll
    for (int j = 0; j < 16; ++j) {
      int ka = kk * 32 + hi * 8 + (j & 7) + ((j >> 3) << 4);
      float av = (ka < 240) ? flat[(size_t)(mt * 16 + row) * 240 + ka] : 0.f;
      au[j] = f32_to_bf16(av);
    }
    v16us bu = *(const v16us*)&sB[kk][lane][0];
    acc = __builtin_amdgcn_wmma_f32_16x16x32_bf16(
        false, __builtin_bit_cast(v16bf, au),
        false, __builtin_bit_cast(v16bf, bu),
        (short)0, acc, false, false);
  }
  float bval = (row < 15) ? fcb[row] : 0.f;
#pragma unroll
  for (int v = 0; v < 8; ++v) {
    int M = v + hi * 8;
    if (row < 15) out[(size_t)(mt * 16 + M) * 15 + row] = acc[v] + bval;
  }
}

// =====================================================================
extern "C" void kernel_launch(void* const* d_in, const int* in_sizes, int n_in,
                              void* d_out, int out_size, void* d_ws, size_t ws_size,
                              hipStream_t stream) {
  const float* x    = (const float*)d_in[0];
  const float* w1   = (const float*)d_in[1];
  const float* b1   = (const float*)d_in[2];
  const float* w2   = (const float*)d_in[3];
  const float* b2   = (const float*)d_in[4];
  const float* w3   = (const float*)d_in[5];
  const float* b3   = (const float*)d_in[6];
  const float* rw1  = (const float*)d_in[7];
  const float* rw2  = (const float*)d_in[8];
  const float* fc_w = (const float*)d_in[9];
  const float* fc_b = (const float*)d_in[10];
  float* out = (float*)d_out;

  const int B = in_sizes[0] / (3 * 64 * 64);     // 512

  // workspace layout (fp32 elements)
  float* ws   = (float*)d_ws;
  float* h1   = ws;                                   // B*8*32*32
  float* h2   = h1 + (size_t)B * 8 * 32 * 32;         // B*16*16*16
  float* h3   = h2 + (size_t)B * 16 * 16 * 16;        // B*32*8*8
  float* uh1  = h3 + (size_t)B * 32 * 8 * 8;          // B*32*64*8
  float* v1   = uh1 + (size_t)B * 32 * 64 * 8;        // B*32*8
  float* flat = v1 + (size_t)B * 32 * 8;              // B*240

  // conv stack: one block per image, input staged in LDS via async DMA
  conv_relu_pool<3, 8, 64, 64><<<B, 256, 0, stream>>>(x, w1, b1, h1, B);
  conv_relu_pool<8, 16, 32, 32><<<B, 256, 0, stream>>>(h1, w2, b2, h2, B);
  conv_relu_pool<16, 32, 16, 16><<<B, 256, 0, stream>>>(h2, w3, b3, h3, B);

  // routing 1: u_hat via dense-packed LDS-fed WMMA, then iterations
  {
    int MT = B / 16;                         // batch m-tiles (multiple of 8)
    int waves = 64 * MT;                     // (r, mtile); 16 n-pairs inside
    uhat1_wmma<<<waves / 8, 256, 0, stream>>>(h3, rw1, uh1, MT);
    routing1_iter<<<(B * 32) / 8, 256, 0, stream>>>(uh1, v1);
  }

  // routing 2 fused
  routing2_fused<<<(B * 15) / 8, 256, 0, stream>>>(v1, rw2, flat);

  // FC via WMMA
  fc_wmma<<<(B / 16) / 8, 256, 0, stream>>>(flat, fc_w, fc_b, out);
}
